// MultiHeadedAttention_22136261444055
// MI455X (gfx1250) — matrix operand order repaired and checked by reading, not compiled, not run on hardware
//
#include <hip/hip_runtime.h>

// ---------------------------------------------------------------------------
// CDNA5 (gfx1250) bf16-WMMA multi-head causal attention forward.
// wave32, V_WMMA_F32_16X16X32_BF16 + GLOBAL_LOAD_ASYNC_TO_LDS_B128,
// double-buffered LDS tiles (async copies overlap WMMA compute).
// ---------------------------------------------------------------------------

typedef __bf16 bf16;
typedef __attribute__((ext_vector_type(4)))  __bf16 bf16x4;
typedef __attribute__((ext_vector_type(8)))  __bf16 bf16x8;
typedef __attribute__((ext_vector_type(16))) __bf16 v16bf;
typedef __attribute__((ext_vector_type(8)))  float  v8f;

#define N_EMBEDS 1024
#define N_HEADS  16
#define HEAD_DIM 64
#define SEQ      2048
#define QKV_LD   3072
#define ATT_SCALE 0.125f   // 64^-0.5

__device__ __forceinline__ v8f wmma_bf16(v16bf a, v16bf b, v8f c) {
  // (neg_a, A, neg_b, B, c_mod, C, reuse_a, reuse_b)
  return __builtin_amdgcn_wmma_f32_16x16x32_bf16(false, a, false, b,
                                                 (short)0, c, false, false);
}

// Build a 16-element bf16 fragment from two contiguous 8-element chunks.
__device__ __forceinline__ v16bf frag2(const bf16* p0, const bf16* p1) {
  bf16x8 lo = *(const bf16x8*)p0;
  bf16x8 hi = *(const bf16x8*)p1;
  v16bf r;
#pragma unroll
  for (int i = 0; i < 8; ++i) { r[i] = lo[i]; r[i + 8] = hi[i]; }
  return r;
}

// Low 32 bits of a flat pointer into LDS == LDS byte offset (ISA 10.2).
__device__ __forceinline__ unsigned lds_off(const void* p) {
  return (unsigned)(unsigned long long)p;
}

// Async 16-byte copy global -> LDS (GVS mode), tracked with ASYNCcnt.
__device__ __forceinline__ void async_b128(unsigned lds_addr,
                                           const bf16* sbase,
                                           unsigned byte_off) {
  asm volatile("global_load_async_to_lds_b128 %0, %1, %2"
               :: "v"(lds_addr), "v"(byte_off), "s"(sbase)
               : "memory");
}
__device__ __forceinline__ void wait_async() {
  asm volatile("s_wait_asynccnt 0" ::: "memory");
}

// ---------------------------------------------------------------------------
// fp32 -> bf16 conversion (vectorized by 4)
// ---------------------------------------------------------------------------
__global__ __launch_bounds__(256)
void cvt_f32_to_bf16(const float* __restrict__ in, bf16* __restrict__ out, int n4) {
  int i = blockIdx.x * 256 + threadIdx.x;
  if (i < n4) {
    float4 v = ((const float4*)in)[i];
    bf16x4 o;
    o[0] = (bf16)v.x; o[1] = (bf16)v.y; o[2] = (bf16)v.z; o[3] = (bf16)v.w;
    ((bf16x4*)out)[i] = o;
  }
}

// ---------------------------------------------------------------------------
// bf16 WMMA GEMM:  C[M,N] = A[M,K] @ B[K,N]  (+bias, fp32 out optional)
// 256 threads = 8 waves (2x4), block tile 64M x 256N, 32x64 per wave,
// K-step 32, double-buffered LDS.
// ---------------------------------------------------------------------------
template <bool OUT_F32, bool BIAS>
__global__ __launch_bounds__(256)
void gemm_bf16_wmma(const bf16* __restrict__ A, const bf16* __restrict__ B,
                    const float* __restrict__ bias, void* __restrict__ Cout,
                    int M, int N, int K) {
  __shared__ __align__(16) bf16 As[2][64 * 40];   // 64 rows x 32 K (stride 40)
  __shared__ __align__(16) bf16 Bt[2][256 * 40];  // transposed: 256 N x 32 K

  const int t    = threadIdx.x;
  const int wave = t >> 5, lane = t & 31;
  const int half = lane >> 4, l15 = lane & 15;
  const int wm   = wave >> 2, wn = wave & 3;
  const int m0   = blockIdx.y * 64;
  const int n0   = blockIdx.x * 256;

  const v8f z = {0.f, 0.f, 0.f, 0.f, 0.f, 0.f, 0.f, 0.f};
  v8f acc[2][4];
#pragma unroll
  for (int i = 0; i < 2; ++i)
#pragma unroll
    for (int j = 0; j < 4; ++j) acc[i][j] = z;

  const int arow = t >> 2, akc = (t & 3) * 8;   // A-tile copy slot
  const int bkr = t >> 4, bnc = (t & 15) * 16;  // B-tile copy slot
  const unsigned a_lds[2] = {
      lds_off(&As[0][0]) + (unsigned)(arow * 40 + akc) * 2,
      lds_off(&As[1][0]) + (unsigned)(arow * 40 + akc) * 2};

  auto load_tile = [&](int buf, int k0) {
    // A tile 64x32: async straight copy, 16B per thread
    async_b128(a_lds[buf], A, (unsigned)((size_t)(m0 + arow) * K + k0 + akc) * 2);
    // B tile 32x256: VGPR load + transposed scatter into LDS
    bf16* bt = &Bt[buf][0];
#pragma unroll
    for (int c = 0; c < 2; ++c) {
      int kk = bkr + c * 16;
#pragma unroll
      for (int j = 0; j < 2; ++j) {
        bf16x8 v = *(const bf16x8*)(B + (size_t)(k0 + kk) * N + n0 + bnc + j * 8);
#pragma unroll
        for (int i = 0; i < 8; ++i) bt[(bnc + j * 8 + i) * 40 + kk] = v[i];
      }
    }
  };

  load_tile(0, 0);
  wait_async();
  __syncthreads();

  const int nk = K >> 5;
  for (int ki = 0; ki < nk; ++ki) {
    const int cur = ki & 1;
    if (ki + 1 < nk) load_tile(1 - cur, (ki + 1) << 5);  // prefetch next

    const bf16* as = &As[cur][0];
    const bf16* bt = &Bt[cur][0];
    v16bf afr[2], bfr[4];
#pragma unroll
    for (int tm = 0; tm < 2; ++tm) {
      const bf16* base = as + (wm * 32 + tm * 16 + l15) * 40;
      afr[tm] = frag2(base + half * 8, base + 16 + half * 8);
    }
#pragma unroll
    for (int tn = 0; tn < 4; ++tn) {
      const bf16* base = bt + (wn * 64 + tn * 16 + l15) * 40 + half * 8;
      bfr[tn] = frag2(base, base + 16);
    }
#pragma unroll
    for (int tm = 0; tm < 2; ++tm)
#pragma unroll
      for (int tn = 0; tn < 4; ++tn)
        acc[tm][tn] = wmma_bf16(afr[tm], bfr[tn], acc[tm][tn]);

    wait_async();     // next-buffer async copies landed
    __syncthreads();  // all waves' writes to next buffer visible
  }

  // Epilogue: C layout -> row = r + 8*half, col = l15
#pragma unroll
  for (int tm = 0; tm < 2; ++tm)
#pragma unroll
    for (int tn = 0; tn < 4; ++tn)
#pragma unroll
      for (int r = 0; r < 8; ++r) {
        int row = m0 + wm * 32 + tm * 16 + r + 8 * half;
        int col = n0 + wn * 64 + tn * 16 + l15;
        float v = acc[tm][tn][r];
        if (OUT_F32) {
          if (BIAS) v += bias[col];
          ((float*)Cout)[(size_t)row * N + col] = v;
        } else {
          ((bf16*)Cout)[(size_t)row * N + col] = (bf16)v;
        }
      }
}

// ---------------------------------------------------------------------------
// Flash attention: grid(x=T/128 query blocks, y=B*H). 8 waves/block,
// 16 query rows per wave, full head_dim=64 accumulators, key tiles of 32,
// double-buffered K/V tiles. Row-sum of P via WMMA against all-ones B.
// ---------------------------------------------------------------------------
__global__ __launch_bounds__(256)
void attn_fwd(const bf16* __restrict__ qkv, bf16* __restrict__ attout) {
  __shared__ __align__(16) bf16 Ks[2][32 * 72];   // 32 keys x 64 d (stride 72)
  __shared__ __align__(16) bf16 Vt[2][64 * 40];   // transposed: 64 d x 32 keys
  __shared__ __align__(16) bf16 Pb[8 * 16 * 32];  // per-wave P bounce buffer

  const int t    = threadIdx.x;
  const int wave = t >> 5, lane = t & 31;
  const int half = lane >> 4, l15 = lane & 15;
  const int qblock = blockIdx.x;
  const int bh = blockIdx.y;
  const int b = bh >> 4, h = bh & 15;
  const int qbase = qblock * 128 + wave * 16;
  bf16* pb = Pb + wave * (16 * 32);

  // Q fragments: two K-steps (d = 0..31, 32..63)
  v16bf qa[2];
  {
    const bf16* qrow =
        qkv + (size_t)(b * SEQ + qbase + l15) * QKV_LD + h * HEAD_DIM;
#pragma unroll
    for (int ks = 0; ks < 2; ++ks)
      qa[ks] = frag2(qrow + ks * 32 + half * 8, qrow + ks * 32 + 16 + half * 8);
  }
  v16bf ones;
#pragma unroll
  for (int i = 0; i < 16; ++i) ones[i] = (bf16)1.0f;

  const v8f z = {0.f, 0.f, 0.f, 0.f, 0.f, 0.f, 0.f, 0.f};
  v8f o[4], lacc = z;
#pragma unroll
  for (int i = 0; i < 4; ++i) o[i] = z;
  float mrow[8];
#pragma unroll
  for (int r = 0; r < 8; ++r) mrow[r] = -3.0e38f;

  const int kr = t >> 3, dc = (t & 7) * 8;  // K/V cooperative-copy slot
  const unsigned k_lds[2] = {
      lds_off(&Ks[0][0]) + (unsigned)(kr * 72 + dc) * 2,
      lds_off(&Ks[1][0]) + (unsigned)(kr * 72 + dc) * 2};

  auto load_kv = [&](int buf, int kt) {
    size_t rowb = (size_t)(b * SEQ + kt * 32 + kr) * QKV_LD + h * HEAD_DIM + dc;
    // K tile (32x64): async straight copy
    async_b128(k_lds[buf], qkv, (unsigned)(rowb + N_EMBEDS) * 2);
    // V tile: VGPR load + transposed scatter (64 d x 32 keys)
    bf16x8 vv = *(const bf16x8*)(qkv + rowb + 2 * N_EMBEDS);
    bf16* vt = &Vt[buf][0];
#pragma unroll
    for (int i = 0; i < 8; ++i) vt[(dc + i) * 40 + kr] = vv[i];
  };

  const int nkt = qblock * 4 + 4;  // causal: key tiles needed by this block
  load_kv(0, 0);
  wait_async();
  __syncthreads();

  for (int kt = 0; kt < nkt; ++kt) {
    const int cur = kt & 1;
    if (kt + 1 < nkt) load_kv(1 - cur, kt + 1);  // prefetch next K/V tile

    if (kt * 32 <= qbase + 15) {  // wave-uniform causal skip
      const bf16* ks = &Ks[cur][0];
      const bf16* vt = &Vt[cur][0];
      // ---- S = Q K^T  (two 16-key halves) ----
      v8f s[2];
#pragma unroll
      for (int nt = 0; nt < 2; ++nt) {
        v8f a = z;
#pragma unroll
        for (int kk = 0; kk < 2; ++kk) {
          const bf16* kb = ks + (nt * 16 + l15) * 72 + kk * 32 + half * 8;
          a = wmma_bf16(qa[kk], frag2(kb, kb + 16), a);
        }
        s[nt] = a;
      }
      // ---- scale ----
#pragma unroll
      for (int nt = 0; nt < 2; ++nt)
#pragma unroll
        for (int r = 0; r < 8; ++r) s[nt][r] = s[nt][r] * ATT_SCALE;
      // ---- causal mask: only tiles straddling the diagonal ----
      if (kt * 32 + 31 > qbase) {
#pragma unroll
        for (int nt = 0; nt < 2; ++nt) {
          int colg = kt * 32 + nt * 16 + l15;
#pragma unroll
          for (int r = 0; r < 8; ++r)
            if (colg > qbase + r + 8 * half) s[nt][r] = -1.0e30f;
        }
      }
      // ---- online softmax: row max via 16-lane shfl_xor, rescale O and l ----
      float p0[8], p1[8];
#pragma unroll
      for (int r = 0; r < 8; ++r) {
        float rm = fmaxf(s[0][r], s[1][r]);
#pragma unroll
        for (int ms = 1; ms < 16; ms <<= 1)
          rm = fmaxf(rm, __shfl_xor(rm, ms, 32));
        float mnew  = fmaxf(mrow[r], rm);
        float alpha = __expf(mrow[r] - mnew);
        p0[r] = __expf(s[0][r] - mnew);
        p1[r] = __expf(s[1][r] - mnew);
        mrow[r] = mnew;
        lacc[r] = lacc[r] * alpha;
#pragma unroll
        for (int dt = 0; dt < 4; ++dt) o[dt][r] = o[dt][r] * alpha;
      }
      // ---- transpose P (C-layout) -> A-layout via per-wave LDS bounce ----
#pragma unroll
      for (int r = 0; r < 8; ++r) {
        int row = r + 8 * half;
        pb[row * 32 + l15]      = (bf16)p0[r];
        pb[row * 32 + 16 + l15] = (bf16)p1[r];
      }
      asm volatile("" ::: "memory");  // DS is in-order within a wave
      const bf16* pr = pb + l15 * 32;
      v16bf pa = frag2(pr + half * 8, pr + 16 + half * 8);
      // ---- row sums: l += P @ ones ----
      lacc = wmma_bf16(pa, ones, lacc);
      // ---- O += P V ----
#pragma unroll
      for (int dt = 0; dt < 4; ++dt) {
        const bf16* vb = vt + (dt * 16 + l15) * 40 + half * 8;
        o[dt] = wmma_bf16(pa, frag2(vb, vb + 16), o[dt]);
      }
    }

    wait_async();     // next-buffer async copies landed
    __syncthreads();  // all waves' writes to next buffer visible
  }

  // ---- normalize + write [B,T,H,hd] (== [B,T,D] flat) ----
#pragma unroll
  for (int dt = 0; dt < 4; ++dt)
#pragma unroll
    for (int r = 0; r < 8; ++r) {
      int row = qbase + r + 8 * half;
      int d   = dt * 16 + l15;
      attout[(size_t)(b * SEQ + row) * N_EMBEDS + h * HEAD_DIM + d] =
          (bf16)(o[dt][r] / lacc[r]);
    }
}

// ---------------------------------------------------------------------------
// Launch
// ---------------------------------------------------------------------------
extern "C" void kernel_launch(void* const* d_in, const int* in_sizes, int n_in,
                              void* d_out, int out_size, void* d_ws,
                              size_t ws_size, hipStream_t stream) {
  const float* x     = (const float*)d_in[0];  // [2,2048,1024]
  const float* w_qkv = (const float*)d_in[1];  // [1024,3072]
  const float* w_prj = (const float*)d_in[2];  // [1024,1024]
  const float* b_prj = (const float*)d_in[3];  // [1024]
  float* out = (float*)d_out;                  // [2,2048,1024] fp32
  char* ws = (char*)d_ws;

  // Workspace layout (bytes)
  bf16* xbf   = (bf16*)(ws + 0);         //  8,388,608  x bf16
  bf16* wqbf  = (bf16*)(ws + 8388608);   //  6,291,456  w_qkv bf16
  bf16* wpbf  = (bf16*)(ws + 14680064);  //  2,097,152  w_proj bf16
  bf16* qkvbf = (bf16*)(ws + 16777216);  // 25,165,824  qkv bf16
  bf16* attbf = (bf16*)(ws + 41943040);  //  8,388,608  attention out bf16

  cvt_f32_to_bf16<<<4096, 256, 0, stream>>>(x, xbf, 4194304 / 4);
  cvt_f32_to_bf16<<<3072, 256, 0, stream>>>(w_qkv, wqbf, 3145728 / 4);
  cvt_f32_to_bf16<<<1024, 256, 0, stream>>>(w_prj, wpbf, 1048576 / 4);

  // qkv = x @ w_qkv : [4096,1024] x [1024,3072] -> bf16
  gemm_bf16_wmma<false, false><<<dim3(12, 64), 256, 0, stream>>>(
      xbf, wqbf, nullptr, (void*)qkvbf, 4096, 3072, 1024);

  // flash attention -> attbf [4096,1024] bf16
  attn_fwd<<<dim3(16, 32), 256, 0, stream>>>(qkvbf, attbf);

  // out = att @ w_proj + b_proj : fp32
  gemm_bf16_wmma<true, true><<<dim3(4, 64), 256, 0, stream>>>(
      attbf, wpbf, b_prj, (void*)out, 4096, 1024, 1024);
}
